// Attention_5334349382130
// MI455X (gfx1250) — matrix operand order repaired and checked by reading, not compiled, not run on hardware
//
#include <hip/hip_runtime.h>
#include <hip/hip_bf16.h>

// ---------------------------------------------------------------------------
// MI455X (gfx1250) multi-head attention.
//   B=4, N=4096, C=256, heads=4, dim_head=64, scale = 1/4096.
// f16 WMMA (v_wmma_f32_16x16x32_f16) with f32 accumulation everywhere.
// Attention stages K/V tiles into LDS with GLOBAL_LOAD_ASYNC_TO_LDS_B128
// (ASYNCcnt-tracked, double buffered) so loads overlap WMMA+softmax compute.
// Pipeline: [cvt/transpose] -> [QKV gemm] -> [flash attention] -> [out gemm].
// Workspace (f16): xh 8MB + 4 W^T 0.5MB + Q 8MB + K 8MB + V^T 8MB + AO 8MB.
// ---------------------------------------------------------------------------

typedef __attribute__((ext_vector_type(16))) _Float16 v16h;
typedef __attribute__((ext_vector_type(8)))  _Float16 v8h;
typedef __attribute__((ext_vector_type(8)))  float    v8f;

#define DEVFN static __device__ __forceinline__

DEVFN v8f wmma_f16f32(v16h a, v16h b, v8f c) {
  // (neg_a, A, neg_b, B, c_mod, C, reuse_a, reuse_b)
  return __builtin_amdgcn_wmma_f32_16x16x32_f16(false, a, false, b,
                                                (short)0, c, false, false);
}

// A operand: 16(M) x 32(K) f16 tile, row-major source, leading dim ld (elems).
// ISA 7.12.2: lane L holds row L%16; lanes 0-15 K-base 0, lanes 16-31 K-base 8;
// a[0..7] = K base..base+7, a[8..15] = K 16+base..16+base+7.
DEVFN v16h load_a16x32(const _Float16* base, int ld) {
  const int lane = threadIdx.x & 31;
  const int row  = lane & 15;
  const int g    = lane >> 4;
  const _Float16* p = base + row * ld + g * 8;
  v8h lo = *(const v8h*)(p);
  v8h hi = *(const v8h*)(p + 16);
  v16h a;
#pragma unroll
  for (int i = 0; i < 8; ++i) { a[i] = lo[i]; a[i + 8] = hi[i]; }
  return a;
}

// B operand: 32(K) x 16(N) f16 tile read from TRANSPOSED storage baseT laid
// out [col][k] with leading dim ld. Lane L holds col L%16; K-base 8*(L/16);
// b[0..7] = K base..base+7, b[8..15] = K 16+base..16+base+7 (matches A).
DEVFN v16h load_b32x16_T(const _Float16* baseT, int ld) {
  const int lane = threadIdx.x & 31;
  const int col  = lane & 15;
  const int g    = lane >> 4;
  const _Float16* p = baseT + col * ld + g * 8;
  v8h lo = *(const v8h*)(p);
  v8h hi = *(const v8h*)(p + 16);
  v16h b;
#pragma unroll
  for (int i = 0; i < 8; ++i) { b[i] = lo[i]; b[i + 8] = hi[i]; }
  return b;
}

// Async global->LDS copy, 16B per lane, tracked by ASYNCcnt (CDNA5 ISA).
// GV addressing: per-lane 64-bit global address, per-lane 32-bit LDS offset.
DEVFN void async_b128(const _Float16* g, const _Float16* l) {
  const unsigned lds_off = (unsigned)(uintptr_t)l;  // low 32b = LDS byte offset
  asm volatile("global_load_async_to_lds_b128 %0, %1, off"
               :: "v"(lds_off), "v"(g) : "memory");
}
DEVFN void wait_async0(void) {
  asm volatile("s_wait_asynccnt 0x0" ::: "memory");
}

// ---------------------------------------------------------------------------
// Conversion kernels
// ---------------------------------------------------------------------------
__global__ void cvt_f32_to_f16(const float* __restrict__ src,
                               _Float16* __restrict__ dst, int n) {
  int i = blockIdx.x * blockDim.x + threadIdx.x;
  if (i < n) dst[i] = (_Float16)src[i];
}

// W[256][256] row-major [k][n]  ->  WT[256][256] = [n][k], f16
__global__ void transpose_w_f16(const float* __restrict__ W,
                                _Float16* __restrict__ WT) {
  int k = blockIdx.x;
  int n = threadIdx.x;
  WT[n * 256 + k] = (_Float16)W[k * 256 + n];
}

// ---------------------------------------------------------------------------
// QKV projection: X[16384,256] @ W[256,256] + b.
// Block = 8 waves; wave owns 16 rows x 64 cols; grid.z selects q/k/v.
// Q,K stored [bh][n][64]; V stored transposed [bh][64][n].
// ---------------------------------------------------------------------------
__global__ void __launch_bounds__(256) qkv_gemm(
    const _Float16* __restrict__ xh,
    const _Float16* __restrict__ WqT, const _Float16* __restrict__ WkT,
    const _Float16* __restrict__ WvT,
    const float* __restrict__ bq, const float* __restrict__ bk,
    const float* __restrict__ bv,
    _Float16* __restrict__ Qh, _Float16* __restrict__ Kh,
    _Float16* __restrict__ Vt) {
  const int lane  = threadIdx.x & 31;
  const int wave  = threadIdx.x >> 5;
  const int col16 = lane & 15;
  const int g     = lane >> 4;
  const int m0    = blockIdx.x * 128 + wave * 16;   // row in [0,16384)
  const int n0    = blockIdx.y * 64;                // col in [0,256)
  const int z     = blockIdx.z;                     // 0=q 1=k 2=v

  const _Float16* WT   = (z == 0) ? WqT : (z == 1) ? WkT : WvT;
  const float*    bias = (z == 0) ? bq  : (z == 1) ? bk  : bv;

  v8f acc[4] = {};
#pragma unroll
  for (int k = 0; k < 256; k += 32) {
    v16h a = load_a16x32(xh + (size_t)m0 * 256 + k, 256);
#pragma unroll
    for (int ns = 0; ns < 4; ++ns) {
      v16h b = load_b32x16_T(WT + (size_t)(n0 + ns * 16) * 256 + k, 256);
      acc[ns] = wmma_f16f32(a, b, acc[ns]);
    }
  }

  const int bidx  = m0 >> 12;      // batch
  const int nbase = m0 & 4095;     // seq position base
#pragma unroll
  for (int ns = 0; ns < 4; ++ns) {
    const int colg = n0 + ns * 16 + col16;
    const int h = colg >> 6, d = colg & 63;
    const float bb = bias[colg];
    const size_t hb = (size_t)(bidx * 4 + h);
#pragma unroll
    for (int r = 0; r < 8; ++r) {
      const int nrow = nbase + r + 8 * g;
      const _Float16 val = (_Float16)(acc[ns][r] + bb);
      if (z == 0)      Qh[(hb * 4096 + nrow) * 64 + d] = val;
      else if (z == 1) Kh[(hb * 4096 + nrow) * 64 + d] = val;
      else             Vt[(hb * 64 + d) * 4096 + nrow] = val;
    }
  }
}

// ---------------------------------------------------------------------------
// Flash attention. Block = 8 waves; each wave owns 16 query rows; the block
// cooperatively streams 64-key chunks of K and V^T through double-buffered
// LDS via async copies. grid = (4096/128, 16 bh-pairs).
// Per wave per chunk: 8 WMMA (S=QK^T) + 8 WMMA (acc+=PV).
// ---------------------------------------------------------------------------
__global__ void __launch_bounds__(256) attn_kernel(
    const _Float16* __restrict__ Qh, const _Float16* __restrict__ Kh,
    const _Float16* __restrict__ Vt, _Float16* __restrict__ AO) {
  __shared__ __align__(16) _Float16 sK[2][64 * 64];   // [key][d]   8KB each
  __shared__ __align__(16) _Float16 sV[2][64 * 64];   // [d][key]   8KB each
  __shared__ __align__(16) _Float16 sP[8][16 * 64];   // per-wave P 2KB each

  const int tid   = threadIdx.x;
  const int lane  = tid & 31;
  const int wave  = tid >> 5;
  const int col16 = lane & 15;
  const int g     = lane >> 4;
  const int bh    = blockIdx.y;                    // 0..15
  const int row0  = blockIdx.x * 128 + wave * 16;  // query row in [0,4096)

  const _Float16* Qb = Qh + (size_t)bh * 4096 * 64;
  const _Float16* Kb = Kh + (size_t)bh * 4096 * 64;
  const _Float16* Vb = Vt + (size_t)bh * 64 * 4096;

  // 16KB chunk = 1024 x 16B slices; 256 threads -> 4 async instrs per thread.
  auto issue_chunk = [&](int buf, int kk) {
#pragma unroll
    for (int i = 0; i < 2; ++i) {        // K tile: contiguous 8KB
      const int ci = i * 256 + tid;      // 16B slice id, 0..511
      async_b128(Kb + (size_t)kk * 64 + ci * 8, &sK[buf][ci * 8]);
    }
#pragma unroll
    for (int i = 0; i < 2; ++i) {        // V tile: 64 rows x 128B, stride 8KB
      const int ci = i * 256 + tid;
      const int d = ci >> 3, pp = ci & 7;
      async_b128(Vb + (size_t)d * 4096 + kk + pp * 8, &sV[buf][d * 64 + pp * 8]);
    }
  };

  // Query A-operands resident for the whole key loop (d = 0..31, 32..63).
  const v16h qa0 = load_a16x32(Qb + (size_t)row0 * 64 + 0,  64);
  const v16h qa1 = load_a16x32(Qb + (size_t)row0 * 64 + 32, 64);

  v8f acc[4] = {};
  float m[8], lsum[8];
#pragma unroll
  for (int r = 0; r < 8; ++r) { m[r] = -__builtin_huge_valf(); lsum[r] = 0.f; }

  const float scale = 1.0f / 4096.0f;    // 1/d^2, exactly 2^-12
  _Float16* pw = &sP[wave][0];

  issue_chunk(0, 0);                     // prologue

  for (int k0 = 0; k0 < 4096; k0 += 64) {
    const int cur = (k0 >> 6) & 1;

    wait_async0();          // own slices of chunk `cur` landed
    __syncthreads();        // everyone's slices landed; prev reads all done
    if (k0 + 64 < 4096) issue_chunk(cur ^ 1, k0 + 64);   // overlap next chunk

    const _Float16* cK = &sK[cur][0];
    const _Float16* cV = &sV[cur][0];

    // ---- S = Q @ K^T : 16 rows x 64 keys (four 16x16 C tiles) ----
    v8f s[4] = {};
#pragma unroll
    for (int kn = 0; kn < 4; ++kn) {
      s[kn] = wmma_f16f32(qa0, load_b32x16_T(cK + kn * 16 * 64 + 0,  64), s[kn]);
      s[kn] = wmma_f16f32(qa1, load_b32x16_T(cK + kn * 16 * 64 + 32, 64), s[kn]);
    }

    // ---- online softmax (row r+8g lives across the 16 lanes of half g) ----
    float rmax[8];
#pragma unroll
    for (int r = 0; r < 8; ++r) {
      float t0 = 0.f, t1 = 0.f;
#pragma unroll
      for (int kn = 0; kn < 4; ++kn) s[kn][r] = s[kn][r] * scale;
      t0 = fmaxf(s[0][r], s[1][r]);
      t1 = fmaxf(s[2][r], s[3][r]);
      rmax[r] = fmaxf(t0, t1);
    }
#pragma unroll
    for (int off = 1; off < 16; off <<= 1) {
#pragma unroll
      for (int r = 0; r < 8; ++r)
        rmax[r] = fmaxf(rmax[r], __shfl_xor(rmax[r], off, 32));
    }

    float rs[8];
#pragma unroll
    for (int r = 0; r < 8; ++r) {
      const float mn    = fmaxf(m[r], rmax[r]);
      const float alpha = __expf(m[r] - mn);
      m[r] = mn;
#pragma unroll
      for (int kn = 0; kn < 4; ++kn) s[kn][r] = __expf(s[kn][r] - mn);
      lsum[r] = lsum[r] * alpha;
#pragma unroll
      for (int t = 0; t < 4; ++t) acc[t][r] = acc[t][r] * alpha;
      rs[r] = (s[0][r] + s[1][r]) + (s[2][r] + s[3][r]);
    }
#pragma unroll
    for (int off = 1; off < 16; off <<= 1) {
#pragma unroll
      for (int r = 0; r < 8; ++r) rs[r] += __shfl_xor(rs[r], off, 32);
    }
#pragma unroll
    for (int r = 0; r < 8; ++r) lsum[r] += rs[r];

    // ---- C-layout -> A-layout relayout of P through per-wave LDS ----
#pragma unroll
    for (int r = 0; r < 8; ++r) {
      const int row = r + 8 * g;
#pragma unroll
      for (int kn = 0; kn < 4; ++kn)
        pw[row * 64 + kn * 16 + col16] = (_Float16)s[kn][r];
    }
    const v16h pa0 = load_a16x32(pw, 64);       // keys 0..31
    const v16h pa1 = load_a16x32(pw + 32, 64);  // keys 32..63

    // ---- acc += P @ V  (V tile in LDS as [d][key]) ----
#pragma unroll
    for (int t = 0; t < 4; ++t) {
      acc[t] = wmma_f16f32(pa0, load_b32x16_T(cV + t * 16 * 64 + 0,  64), acc[t]);
      acc[t] = wmma_f16f32(pa1, load_b32x16_T(cV + t * 16 * 64 + 32, 64), acc[t]);
    }
  }

  // ---- epilogue: normalize, write AO[b][n][h*64+d] (f16) ----
  const int b = bh >> 2, h = bh & 3;
#pragma unroll
  for (int r = 0; r < 8; ++r) {
    const float inv = 1.0f / lsum[r];
    const int n = row0 + r + 8 * g;
#pragma unroll
    for (int t = 0; t < 4; ++t) {
      const int d = t * 16 + col16;
      AO[((size_t)(b * 4096 + n)) * 256 + h * 64 + d] =
          (_Float16)(acc[t][r] * inv);
    }
  }
}

// ---------------------------------------------------------------------------
// Output projection: AO[16384,256] @ Wo[256,256] + bo -> f32 out.
// ---------------------------------------------------------------------------
__global__ void __launch_bounds__(256) out_proj_gemm(
    const _Float16* __restrict__ AO, const _Float16* __restrict__ WoT,
    const float* __restrict__ bo, float* __restrict__ out) {
  const int lane  = threadIdx.x & 31;
  const int wave  = threadIdx.x >> 5;
  const int col16 = lane & 15;
  const int g     = lane >> 4;
  const int m0    = blockIdx.x * 128 + wave * 16;
  const int n0    = blockIdx.y * 64;

  v8f acc[4] = {};
#pragma unroll
  for (int k = 0; k < 256; k += 32) {
    v16h a = load_a16x32(AO + (size_t)m0 * 256 + k, 256);
#pragma unroll
    for (int ns = 0; ns < 4; ++ns) {
      v16h b = load_b32x16_T(WoT + (size_t)(n0 + ns * 16) * 256 + k, 256);
      acc[ns] = wmma_f16f32(a, b, acc[ns]);
    }
  }
#pragma unroll
  for (int ns = 0; ns < 4; ++ns) {
    const int colg = n0 + ns * 16 + col16;
    const float bb = bo[colg];
#pragma unroll
    for (int r = 0; r < 8; ++r) {
      const int row = m0 + r + 8 * g;
      out[(size_t)row * 256 + colg] = acc[ns][r] + bb;   // coalesced b32
    }
  }
}

// ---------------------------------------------------------------------------
// Host-side launcher
// ---------------------------------------------------------------------------
extern "C" void kernel_launch(void* const* d_in, const int* in_sizes, int n_in,
                              void* d_out, int out_size, void* d_ws,
                              size_t ws_size, hipStream_t stream) {
  (void)in_sizes; (void)n_in; (void)out_size; (void)ws_size;

  const float* x  = (const float*)d_in[0];
  const float* Wq = (const float*)d_in[1];
  const float* bq = (const float*)d_in[2];
  const float* Wk = (const float*)d_in[3];
  const float* bk = (const float*)d_in[4];
  const float* Wv = (const float*)d_in[5];
  const float* bv = (const float*)d_in[6];
  const float* Wo = (const float*)d_in[7];
  const float* bo = (const float*)d_in[8];
  float* out = (float*)d_out;

  const size_t BN  = 16384;       // B*N
  const size_t C   = 256;
  const size_t XE  = BN * C;      // 4,194,304 elems
  const size_t WE  = C * C;       // 65,536 elems

  _Float16* ws  = (_Float16*)d_ws;
  _Float16* xh  = ws;
  _Float16* WqT = xh  + XE;
  _Float16* WkT = WqT + WE;
  _Float16* WvT = WkT + WE;
  _Float16* WoT = WvT + WE;
  _Float16* Qh  = WoT + WE;       // [16][4096][64]
  _Float16* Kh  = Qh  + XE;       // [16][4096][64]
  _Float16* Vt  = Kh  + XE;       // [16][64][4096]
  _Float16* AO  = Vt  + XE;       // [16384][256]

  cvt_f32_to_f16<<<(int)((XE + 255) / 256), 256, 0, stream>>>(x, xh, (int)XE);
  transpose_w_f16<<<256, 256, 0, stream>>>(Wq, WqT);
  transpose_w_f16<<<256, 256, 0, stream>>>(Wk, WkT);
  transpose_w_f16<<<256, 256, 0, stream>>>(Wv, WvT);
  transpose_w_f16<<<256, 256, 0, stream>>>(Wo, WoT);

  qkv_gemm<<<dim3(128, 4, 3), 256, 0, stream>>>(xh, WqT, WkT, WvT,
                                                bq, bk, bv, Qh, Kh, Vt);

  attn_kernel<<<dim3(32, 16), 256, 0, stream>>>(Qh, Kh, Vt, AO);

  out_proj_gemm<<<dim3(128, 4), 256, 0, stream>>>(AO, WoT, bo, out);
}
